// Resblock_con5_with_BNTT_25254407700788
// MI455X (gfx1250) — matrix operand order repaired and checked by reading, not compiled, not run on hardware
//
#include <hip/hip_runtime.h>

typedef __attribute__((ext_vector_type(16))) _Float16 v16h;
typedef __attribute__((ext_vector_type(8)))  float    v8f;

#define T_  4
#define B_  8
#define C_  32
#define H_  128
#define W_  128
#define HW_ (H_*W_)
#define NIMG_ (T_*B_)          // 32 images of [128,128,32]
#define EPS_ 1e-5f
#define NBLK_ (NIMG_*H_)       // 4096 conv workgroups (one per image row)

// ---------------------------------------------------------------------------
// NCHW f32 -> NHWC f16 transpose (32x32 channel/width tiles through LDS)
// ---------------------------------------------------------------------------
__global__ __launch_bounds__(1024)
void nchw_to_nhwc_h(const float* __restrict__ x, _Float16* __restrict__ xh) {
    __shared__ float t[32][33];
    const int w0  = blockIdx.x * 32;
    const int h   = blockIdx.y;
    const int img = blockIdx.z;
    const int tx = threadIdx.x, ty = threadIdx.y;
    // read x[img, c=ty, h, w0+tx] (coalesced along w)
    t[ty][tx] = x[(((size_t)img * C_ + ty) * H_ + h) * W_ + (w0 + tx)];
    __syncthreads();
    // write xh[img, h, w0+ty, c=tx] (coalesced along c)
    xh[(((size_t)img * H_ + h) * W_ + (w0 + ty)) * C_ + tx] = (_Float16)t[tx][ty];
}

// ---------------------------------------------------------------------------
// Repack OIHW f32 weights into per-tap WMMA B fragments (f16):
// bw[tap][ntile][lane][e]:  col n=lane%16, h=lane/16:
//                           K = 8h + e (e<8), K = 16 + 8h + (e-8) (e>=8)
// ---------------------------------------------------------------------------
__global__ void repack_w(const float* __restrict__ wsrc, _Float16* __restrict__ bw) {
    int idx = blockIdx.x * blockDim.x + threadIdx.x;   // 25*2*32*16 = 25600
    if (idx >= 25 * 2 * 32 * 16) return;
    int e    = idx & 15;
    int lane = (idx >> 4) & 31;
    int nt   = (idx >> 9) & 1;
    int tap  = idx >> 10;
    int kh = tap / 5, kw = tap % 5;
    int o = nt * 16 + (lane & 15);
    int i = (e < 8) ? ((lane >> 4) * 8 + e) : (16 + (lane >> 4) * 8 + (e - 8));
    bw[idx] = (_Float16)wsrc[((o * C_ + i) * 5 + kh) * 5 + kw];
}

// ---------------------------------------------------------------------------
// 5x5 conv as implicit GEMM on the WMMA pipe.
// One workgroup = one (image, row): 8 waves x (16 pixels x 32 out ch).
// Per tap: A = 16x32 f16 pixel/channel tile from LDS, B = 32x16 f16 weights,
// 2x v_wmma_f32_16x16x32_f16 accumulate into f32.
// Also emits deterministic per-block channel sum / sumsq for the TEBN stats.
// ---------------------------------------------------------------------------
__global__ __launch_bounds__(256)
void conv5_wmma(const _Float16* __restrict__ in, const _Float16* __restrict__ bw,
                float* __restrict__ out,
                float* __restrict__ bsum, float* __restrict__ bsq) {
    __shared__ _Float16 tile[5 * 132 * C_];   // rows h-2..h+2, cols -2..129, 32 ch
    __shared__ float redS[32];
    __shared__ float redQ[32];

    const int h   = blockIdx.x;
    const int img = blockIdx.y;
    const int tid = threadIdx.x;

    if (tid < 32) { redS[tid] = 0.f; redQ[tid] = 0.f; }

    // stage halo tile (zero pad outside the image)
    for (int slot = tid; slot < 5 * 132; slot += 256) {
        int r = slot / 132, cc = slot % 132;
        int gr = h + r - 2, gc = cc - 2;
        uint4* dst = (uint4*)(tile + slot * C_);
        if (gr >= 0 && gr < H_ && gc >= 0 && gc < W_) {
            const uint4* src =
                (const uint4*)(in + (((size_t)img * H_ + gr) * W_ + gc) * C_);
            dst[0] = src[0]; dst[1] = src[1]; dst[2] = src[2]; dst[3] = src[3];
        } else {
            uint4 z = {0u, 0u, 0u, 0u};
            dst[0] = z; dst[1] = z; dst[2] = z; dst[3] = z;
        }
    }
    __syncthreads();

    const int lane  = tid & 31;
    const int wave  = tid >> 5;
    const int m     = lane & 15;    // pixel row within A tile / output channel col
    const int hi    = lane >> 4;
    const int wbase = wave * 16;    // this wave's 16 output pixels

    v8f acc0 = {};   // out channels 0..15
    v8f acc1 = {};   // out channels 16..31

    for (int kh = 0; kh < 5; ++kh) {
        for (int kw = 0; kw < 5; ++kw) {
            const int tap = kh * 5 + kw;
            // A fragment: pixel (wbase+m) at tap offset; per-lane K halves
            const _Float16* p = tile + (kh * 132 + (wbase + m + kw)) * C_;
            v16h a;
            *(uint4*)&a         = *(const uint4*)(p + hi * 8);        // K 0..7 / 8..15
            *(((uint4*)&a) + 1) = *(const uint4*)(p + 16 + hi * 8);   // K 16..23 / 24..31
            const v16h b0 = *(const v16h*)(bw + (((tap * 2) + 0) * 32 + lane) * 16);
            const v16h b1 = *(const v16h*)(bw + (((tap * 2) + 1) * 32 + lane) * 16);
            acc0 = __builtin_amdgcn_wmma_f32_16x16x32_f16(
                false, a, false, b0, (short)0, acc0, false, false);
            acc1 = __builtin_amdgcn_wmma_f32_16x16x32_f16(
                false, a, false, b1, (short)0, acc1, false, false);
        }
    }

    // Epilogue: store f32 NHWC + per-channel partial stats.
    // D layout: VGPR r -> pixel M = r + hi*8, channel N = m (acc0) / 16+m (acc1)
    float s0 = 0.f, q0 = 0.f, s1 = 0.f, q1 = 0.f;
    const size_t rowbase = ((size_t)img * H_ + h) * W_;
    #pragma unroll
    for (int r = 0; r < 8; ++r) {
        int wp = wbase + hi * 8 + r;
        float v0 = acc0[r], v1 = acc1[r];
        out[(rowbase + wp) * C_ + m]        = v0;
        out[(rowbase + wp) * C_ + (16 + m)] = v1;
        s0 += v0; q0 += v0 * v0;
        s1 += v1; q1 += v1 * v1;
    }
    atomicAdd(&redS[m], s0);        atomicAdd(&redQ[m], q0);
    atomicAdd(&redS[16 + m], s1);   atomicAdd(&redQ[16 + m], q1);
    __syncthreads();

    if (tid < 32) {
        const size_t blk = (size_t)img * H_ + h;
        bsum[blk * 32 + tid] = redS[tid];
        bsq [blk * 32 + tid] = redQ[tid];
    }
}

// ---------------------------------------------------------------------------
// Deterministic reduction of per-block stats -> per-(t,c) affine BN coeffs:
//   ab[t*C + c]        = gamma*rstd*scale[t]
//   ab[T*C + t*C + c]  = (beta - mean*gamma*rstd)*scale[t]
// ---------------------------------------------------------------------------
__global__ __launch_bounds__(256)
void bn_stats(const float* __restrict__ bsum, const float* __restrict__ bsq,
              const float* __restrict__ gamma, const float* __restrict__ beta,
              const float* __restrict__ scale, float* __restrict__ ab) {
    __shared__ float ss[8][32];
    __shared__ float qq[8][32];
    const int c    = threadIdx.x & 31;
    const int part = threadIdx.x >> 5;          // 8 chunks of 512 blocks
    float s = 0.f, q = 0.f;
    for (int bl = part * 512; bl < (part + 1) * 512; ++bl) {
        s += bsum[bl * 32 + c];
        q += bsq [bl * 32 + c];
    }
    ss[part][c] = s; qq[part][c] = q;
    __syncthreads();
    if (threadIdx.x < 32) {
        float S = 0.f, Q = 0.f;
        #pragma unroll
        for (int p = 0; p < 8; ++p) { S += ss[p][c]; Q += qq[p][c]; }
        const float N = (float)((size_t)T_ * B_ * HW_);
        float mean = S / N;
        float var  = Q / N - mean * mean;
        float rstd = rsqrtf(var + EPS_);
        float g = gamma[c] * rstd;
        #pragma unroll
        for (int t = 0; t < T_; ++t) {
            ab[t * C_ + c]           = g * scale[t];
            ab[T_ * C_ + t * C_ + c] = (beta[c] - mean * g) * scale[t];
        }
    }
}

// ---------------------------------------------------------------------------
// BN apply + LIF scan over T (tau=2, v_th=1, hard reset). Spikes -> f16 NHWC.
// ---------------------------------------------------------------------------
__global__ __launch_bounds__(256)
void bn_lif(const float* __restrict__ y, const float* __restrict__ ab,
            _Float16* __restrict__ sp) {
    size_t i = (size_t)blockIdx.x * blockDim.x + threadIdx.x;   // over (b,h,w,c)
    const size_t stride = (size_t)B_ * HW_ * C_;                // per-timestep stride
    if (i >= stride) return;
    const int c = (int)(i & 31);
    float v = 0.f;
    #pragma unroll
    for (int t = 0; t < T_; ++t) {
        float z = ab[t * C_ + c] * y[i + (size_t)t * stride] + ab[T_ * C_ + t * C_ + c];
        v += (z - v) * 0.5f;
        float spike = (v >= 1.f) ? 1.f : 0.f;
        v *= (1.f - spike);
        sp[i + (size_t)t * stride] = (_Float16)spike;
    }
}

// ---------------------------------------------------------------------------
// BN apply + residual + LIF; writes final spikes in NCHW f32 (reference order).
// ---------------------------------------------------------------------------
__global__ __launch_bounds__(256)
void bn_res_lif_out(const float* __restrict__ y, const float* __restrict__ ab,
                    const float* __restrict__ x, float* __restrict__ out) {
    size_t i = (size_t)blockIdx.x * blockDim.x + threadIdx.x;   // over (b,c,h,w)
    const size_t strideT = (size_t)B_ * C_ * HW_;
    if (i >= strideT) return;
    const int w  = (int)(i & 127);
    const int hh = (int)((i >> 7) & 127);
    const int c  = (int)((i >> 14) & 31);
    const int b  = (int)(i >> 19);
    float v = 0.f;
    #pragma unroll
    for (int t = 0; t < T_; ++t) {
        size_t g  = i + (size_t)t * strideT;                            // NCHW
        size_t yi = ((((size_t)(t * B_ + b)) * H_ + hh) * W_ + w) * C_ + c; // NHWC
        float z = ab[t * C_ + c] * y[yi] + ab[T_ * C_ + t * C_ + c] + x[g];
        v += (z - v) * 0.5f;
        float spike = (v >= 1.f) ? 1.f : 0.f;
        v *= (1.f - spike);
        out[g] = spike;
    }
}

// ---------------------------------------------------------------------------
extern "C" void kernel_launch(void* const* d_in, const int* in_sizes, int n_in,
                              void* d_out, int out_size, void* d_ws, size_t ws_size,
                              hipStream_t stream) {
    const float* x   = (const float*)d_in[0];
    const float* w1  = (const float*)d_in[1];
    const float* g1  = (const float*)d_in[2];
    const float* be1 = (const float*)d_in[3];
    const float* sc1 = (const float*)d_in[4];
    const float* w2  = (const float*)d_in[5];
    const float* g2  = (const float*)d_in[6];
    const float* be2 = (const float*)d_in[7];
    const float* sc2 = (const float*)d_in[8];

    char* ws = (char*)d_ws;
    size_t off = 0;
    auto alloc = [&](size_t bytes) -> void* {
        void* p = ws + off;
        off += (bytes + 255) & ~(size_t)255;
        return p;
    };
    const size_t NE = (size_t)T_ * B_ * HW_ * C_;   // 16,777,216 elements

    _Float16* xh   = (_Float16*)alloc(NE * 2);              // NHWC f16 input
    _Float16* sp   = (_Float16*)alloc(NE * 2);              // NHWC f16 spikes
    float*    y    = (float*)   alloc(NE * 4);              // conv out (reused)
    _Float16* bw1  = (_Float16*)alloc(25 * 2 * 32 * 16 * 2);
    _Float16* bw2  = (_Float16*)alloc(25 * 2 * 32 * 16 * 2);
    float*    bs1  = (float*)   alloc((size_t)NBLK_ * 32 * 4);
    float*    bq1  = (float*)   alloc((size_t)NBLK_ * 32 * 4);
    float*    bs2  = (float*)   alloc((size_t)NBLK_ * 32 * 4);
    float*    bq2  = (float*)   alloc((size_t)NBLK_ * 32 * 4);
    float*    ab1  = (float*)   alloc(2 * T_ * C_ * 4);
    float*    ab2  = (float*)   alloc(2 * T_ * C_ * 4);
    (void)ws_size; (void)in_sizes; (void)n_in; (void)out_size;

    nchw_to_nhwc_h<<<dim3(4, 128, 32), dim3(32, 32), 0, stream>>>(x, xh);
    repack_w<<<100, 256, 0, stream>>>(w1, bw1);
    repack_w<<<100, 256, 0, stream>>>(w2, bw2);

    conv5_wmma<<<dim3(H_, NIMG_), 256, 0, stream>>>(xh, bw1, y, bs1, bq1);
    bn_stats<<<1, 256, 0, stream>>>(bs1, bq1, g1, be1, sc1, ab1);
    bn_lif<<<(int)(((size_t)B_ * HW_ * C_) / 256), 256, 0, stream>>>(y, ab1, sp);

    conv5_wmma<<<dim3(H_, NIMG_), 256, 0, stream>>>(sp, bw2, y, bs2, bq2);
    bn_stats<<<1, 256, 0, stream>>>(bs2, bq2, g2, be2, sc2, ab2);
    bn_res_lif_out<<<(int)(((size_t)B_ * C_ * HW_) / 256), 256, 0, stream>>>(
        y, ab2, x, (float*)d_out);
}
